// SelfAttention_13494787244060
// MI455X (gfx1250) — matrix operand order repaired and checked by reading, not compiled, not run on hardware
//
#include <hip/hip_runtime.h>
#include <math.h>

// ---------------------------------------------------------------------------
// Types for CDNA5 WMMA (wave32): v_wmma_f32_16x16x32_bf16
// ---------------------------------------------------------------------------
typedef __attribute__((ext_vector_type(16))) __bf16 v16bf;
typedef __attribute__((ext_vector_type(8)))  __bf16 v8bf;
typedef __attribute__((ext_vector_type(8)))  float  v8f;

#define S_LEN 2048
#define H_DIM 4096
#define NHEAD 32
#define HDIM  128

static __device__ inline v8f zero8() {
  v8f z = {0.f,0.f,0.f,0.f,0.f,0.f,0.f,0.f};
  return z;
}

static __device__ inline v8f wmma_bf16(v16bf a, v16bf b, v8f c) {
  return __builtin_amdgcn_wmma_f32_16x16x32_bf16(false, a, false, b, (short)0, c, false, false);
}

static __device__ inline v16bf comb(v8bf lo, v8bf hi) {
  v16bf r;
#pragma unroll
  for (int i = 0; i < 8; ++i) { r[i] = lo[i]; r[8 + i] = hi[i]; }
  return r;
}

// A-fragment (16x32 bf16, MxK): lanes 0-15 hold K 0-7 & 16-23; lanes 16-31 hold
// K 8-15 & 24-31 (ISA 7.12.2).  rowp points at this lane's row (M = lane%16).
static __device__ inline v16bf ld_afrag(const __bf16* rowp, int k0, int half) {
  v8bf lo = *(const v8bf*)(rowp + k0 + half * 8);
  v8bf hi = *(const v8bf*)(rowp + k0 + 16 + half * 8);
  return comb(lo, hi);
}

// B-fragment (32x16 bf16, KxN) read from an N x K row-major buffer: lane's
// column N = lane%16; lanes 0-15 hold K 0-7 & 16-23, lanes 16-31 hold K 8-15 & 24-31.
static __device__ inline v16bf ld_bfrag(const __bf16* rowp, int k0, int half) {
  v8bf lo = *(const v8bf*)(rowp + k0 + half * 8);
  v8bf hi = *(const v8bf*)(rowp + k0 + 16 + half * 8);
  return comb(lo, hi);
}

// Stage 16 fp32 -> 16 bf16 into LDS (two b128 LDS stores).
static __device__ inline void stage_f32(const float* __restrict__ src, __bf16* dst) {
  const float4* s4 = (const float4*)src;
  float tmp[16];
#pragma unroll
  for (int q = 0; q < 4; ++q) {
    float4 f = s4[q];
    tmp[4*q+0] = f.x; tmp[4*q+1] = f.y; tmp[4*q+2] = f.z; tmp[4*q+3] = f.w;
  }
  v8bf a, b;
#pragma unroll
  for (int e = 0; e < 8; ++e) { a[e] = (__bf16)tmp[e]; b[e] = (__bf16)tmp[8 + e]; }
  *(v8bf*)dst = a;
  *(v8bf*)(dst + 8) = b;
}

// gfx1250 async global -> LDS copy of 16 bytes per lane (ASYNCcnt tracked).
// LDS dest address = low 32 bits of the generic shared pointer.
static __device__ inline void async_copy16B(const __bf16* gsrc, __bf16* ldst) {
  unsigned la = (unsigned)(uintptr_t)ldst;
  asm volatile("global_load_async_to_lds_b128 %0, %1, off"
               :: "v"(la), "v"(gsrc)
               : "memory");
}

// Copy 16 bf16 (32B) global -> LDS via two async b128 transfers.
static __device__ inline void stage_async32B(const __bf16* gsrc, __bf16* ldst) {
  async_copy16B(gsrc, ldst);
  async_copy16B(gsrc + 8, ldst + 8);
}

static __device__ inline void wait_async0() {
  asm volatile("s_wait_asynccnt 0x0" ::: "memory");
}

// ---------------------------------------------------------------------------
// Kernel 1: QKV = hidden @ W_qkv^T + b_qkv, written as bf16 per-head layouts.
// Grid (16, 96), block 256.  Macro tile 128x128, K staged 64 at a time.
// (Both operands need fp32->bf16 conversion, so staging stays on the VALU path.)
// ---------------------------------------------------------------------------
__global__ __launch_bounds__(256) void qkv_gemm_kernel(
    const float* __restrict__ A, const float* __restrict__ W,
    const float* __restrict__ bias,
    __bf16* __restrict__ Qb, __bf16* __restrict__ Kb, __bf16* __restrict__ Vb)
{
  __shared__ alignas(16) __bf16 As[128 * 72];
  __shared__ alignas(16) __bf16 Bs[128 * 72];
  const int tid = threadIdx.x, bm = blockIdx.x, bn = blockIdx.y;
  const int w = tid >> 5, lane = tid & 31, half = lane >> 4, l16 = lane & 15;
  const int wr = w >> 2, wc = w & 3;            // 2x4 wave grid, each wave 64x32

  v8f acc[4][2];
#pragma unroll
  for (int i = 0; i < 4; ++i)
#pragma unroll
    for (int j = 0; j < 2; ++j) acc[i][j] = zero8();

  for (int k0 = 0; k0 < H_DIM; k0 += 64) {
    __syncthreads();
#pragma unroll
    for (int c = tid; c < 512; c += 256) {
      int row = c >> 2, kc = (c & 3) * 16;
      stage_f32(A + (size_t)(bm * 128 + row) * H_DIM + k0 + kc, &As[row * 72 + kc]);
    }
#pragma unroll
    for (int c = tid; c < 512; c += 256) {
      int row = c >> 2, kc = (c & 3) * 16;
      stage_f32(W + (size_t)(bn * 128 + row) * H_DIM + k0 + kc, &Bs[row * 72 + kc]);
    }
    if (k0 + 64 < H_DIM)
      __builtin_prefetch(W + (size_t)(bn * 128 + (tid >> 1)) * H_DIM + k0 + 64, 0, 1);
    __syncthreads();
#pragma unroll
    for (int kk = 0; kk < 64; kk += 32) {
      v16bf af[4];
#pragma unroll
      for (int i = 0; i < 4; ++i)
        af[i] = ld_afrag(&As[(wr * 64 + i * 16 + l16) * 72], kk, half);
      v16bf bfv[2];
#pragma unroll
      for (int j = 0; j < 2; ++j)
        bfv[j] = ld_bfrag(&Bs[(wc * 32 + j * 16 + l16) * 72], kk, half);
#pragma unroll
      for (int i = 0; i < 4; ++i)
#pragma unroll
        for (int j = 0; j < 2; ++j)
          acc[i][j] = wmma_bf16(af[i], bfv[j], acc[i][j]);
    }
  }

  // Epilogue: bias + scatter into Q/K/V [head][s][d] bf16.
#pragma unroll
  for (int j = 0; j < 2; ++j) {
    const int n = bn * 128 + wc * 32 + j * 16 + l16;    // feature in [0,12288)
    const float bv = bias[n];
    const int hh = n / 384, r = n % 384;
    __bf16* dst; int doff;
    if (r < 128)      { dst = Qb; doff = r; }
    else if (r < 256) { dst = Kb; doff = r - 128; }
    else              { dst = Vb; doff = r - 256; }
    const size_t base = (size_t)hh * S_LEN * HDIM + doff;
#pragma unroll
    for (int i = 0; i < 4; ++i) {
#pragma unroll
      for (int v = 0; v < 8; ++v) {
        const int srow = bm * 128 + wr * 64 + i * 16 + half * 8 + v;
        dst[base + (size_t)srow * HDIM] = (__bf16)(acc[i][j][v] + bv);
      }
    }
  }
}

// ---------------------------------------------------------------------------
// Kernel 2: RoPE in place on first ROT=32 dims of Q and K.
// inv_freq[i] = 10000^(-i/16), pairs (d, d+16).
// ---------------------------------------------------------------------------
__global__ __launch_bounds__(256) void rope_kernel(__bf16* __restrict__ Qb,
                                                   __bf16* __restrict__ Kb)
{
  const int idx = blockIdx.x * 256 + threadIdx.x;
  const int i = idx & 15;
  const int h = (idx >> 4) & 31;
  const int s = idx >> 9;
  if (s >= S_LEN) return;
  const float f = (float)s * powf(10000.f, -(float)i * (1.f / 16.f));
  const float c = cosf(f), sn = sinf(f);
  const size_t base = ((size_t)h * S_LEN + s) * HDIM;
  {
    float x1 = (float)Qb[base + i], x2 = (float)Qb[base + i + 16];
    Qb[base + i]      = (__bf16)(x1 * c - x2 * sn);
    Qb[base + i + 16] = (__bf16)(x2 * c + x1 * sn);
  }
  {
    float x1 = (float)Kb[base + i], x2 = (float)Kb[base + i + 16];
    Kb[base + i]      = (__bf16)(x1 * c - x2 * sn);
    Kb[base + i + 16] = (__bf16)(x2 * c + x1 * sn);
  }
}

// ---------------------------------------------------------------------------
// Kernel 3: causal flash attention.  Grid (S/128, NH), block 256 (8 waves).
// Each wave owns 16 query rows -> softmax stats stay inside the wave.
// Q/K blocks staged via gfx1250 async global->LDS DMA (ASYNCcnt).
// ---------------------------------------------------------------------------
__global__ __launch_bounds__(256) void attn_kernel(
    const __bf16* __restrict__ Qb, const __bf16* __restrict__ Kb,
    const __bf16* __restrict__ Vb, __bf16* __restrict__ Ctx)
{
  __shared__ alignas(16) __bf16 Qs[128 * 136];
  __shared__ alignas(16) __bf16 Ks[64 * 136];
  __shared__ alignas(16) __bf16 Vt[128 * 72];   // V transposed: [dim][key]
  __shared__ alignas(16) __bf16 Ps[128 * 72];   // per-wave P re-fragment buffer

  const int tid = threadIdx.x;
  const int h = blockIdx.y;
  const int q0 = blockIdx.x * 128;
  const int w = tid >> 5, lane = tid & 31, half = lane >> 4, l16 = lane & 15;
  const size_t headBase = (size_t)h * S_LEN * HDIM;

  // Async-stage the 128x128 Q block once (waited before first compute barrier).
  for (int c = tid; c < 1024; c += 256) {
    int row = c >> 3, dc = (c & 7) * 16;
    stage_async32B(Qb + headBase + (size_t)(q0 + row) * HDIM + dc,
                   &Qs[row * 136 + dc]);
  }

  float m[8], l[8];
  v8f O[8];
#pragma unroll
  for (int v = 0; v < 8; ++v) { m[v] = -1e30f; l[v] = 0.f; }
#pragma unroll
  for (int t = 0; t < 8; ++t) O[t] = zero8();

  const int kb_hi = (q0 + 127) >> 6;
  const float scale = 0.08838834764831845f;     // 1/sqrt(128)
  const int qw = q0 + w * 16;

  for (int kb = 0; kb <= kb_hi; ++kb) {
    __syncthreads();
    for (int c = tid; c < 512; c += 256) {      // K block 64x128 (async DMA)
      int row = c >> 3, dc = (c & 7) * 16;
      stage_async32B(Kb + headBase + (size_t)(kb * 64 + row) * HDIM + dc,
                     &Ks[row * 136 + dc]);
    }
    for (int c = tid; c < 1024; c += 256) {     // V block 64x128 -> transposed
      int key = c >> 4, dc = (c & 15) * 8;
      v8bf vv = *(const v8bf*)(Vb + headBase + (size_t)(kb * 64 + key) * HDIM + dc);
#pragma unroll
      for (int e = 0; e < 8; ++e) Vt[(dc + e) * 72 + key] = vv[e];
    }
    wait_async0();                              // async LDS writes visible
    __syncthreads();

    if (kb * 64 <= qw + 15) {                   // wave-uniform: EXEC stays full
      v8f S4[4];
#pragma unroll
      for (int t = 0; t < 4; ++t) S4[t] = zero8();
#pragma unroll
      for (int ks = 0; ks < 4; ++ks) {
        v16bf a = ld_afrag(&Qs[(w * 16 + l16) * 136], ks * 32, half);
#pragma unroll
        for (int t = 0; t < 4; ++t) {
          v16bf b = ld_bfrag(&Ks[(t * 16 + l16) * 136], ks * 32, half);
          S4[t] = wmma_bf16(a, b, S4[t]);
        }
      }
      const bool needMask = (kb * 64 + 63) > qw;
#pragma unroll
      for (int v = 0; v < 8; ++v) {
        const int qrow = qw + half * 8 + v;
        float mx = -1e30f;
#pragma unroll
        for (int t = 0; t < 4; ++t) {
          float s = S4[t][v] * scale;
          if (needMask && (kb * 64 + t * 16 + l16) > qrow) s = -1e30f;
          S4[t][v] = s;
          mx = fmaxf(mx, s);
        }
#pragma unroll
        for (int off = 1; off < 16; off <<= 1)
          mx = fmaxf(mx, __shfl_xor(mx, off, 16));
        const float mn = fmaxf(m[v], mx);
        const float alpha = __expf(m[v] - mn);
        m[v] = mn;
        float rs = 0.f;
#pragma unroll
        for (int t = 0; t < 4; ++t) {
          float p = __expf(S4[t][v] - mn);
          S4[t][v] = p;
          rs += p;
        }
#pragma unroll
        for (int off = 1; off < 16; off <<= 1) rs += __shfl_xor(rs, off, 16);
        l[v] = l[v] * alpha + rs;
#pragma unroll
        for (int ot = 0; ot < 8; ++ot) O[ot][v] *= alpha;
      }
      // P (C-layout) -> LDS -> A-fragment layout for P @ V.
#pragma unroll
      for (int v = 0; v < 8; ++v) {
        __bf16* pr = &Ps[(w * 16 + half * 8 + v) * 72];
#pragma unroll
        for (int t = 0; t < 4; ++t) pr[t * 16 + l16] = (__bf16)S4[t][v];
      }
#pragma unroll
      for (int ks = 0; ks < 2; ++ks) {
        v16bf a = ld_afrag(&Ps[(w * 16 + l16) * 72], ks * 32, half);
#pragma unroll
        for (int ot = 0; ot < 8; ++ot) {
          v16bf b = ld_bfrag(&Vt[(ot * 16 + l16) * 72], ks * 32, half);
          O[ot] = wmma_bf16(a, b, O[ot]);
        }
      }
    }
  }

  // Normalize and write ctx[s][h*128 + d] as bf16.
#pragma unroll
  for (int v = 0; v < 8; ++v) {
    const float inv = 1.f / l[v];
    const size_t srow = (size_t)(qw + half * 8 + v) * H_DIM + (size_t)h * HDIM;
#pragma unroll
    for (int ot = 0; ot < 8; ++ot)
      Ctx[srow + ot * 16 + l16] = (__bf16)(O[ot][v] * inv);
  }
}

// ---------------------------------------------------------------------------
// Kernel 4: out = ctx(bf16) @ W_dense^T + b_dense (fp32 out).  Grid (16, 32).
// A (bf16) staged via async global->LDS DMA; W converts fp32->bf16 on the fly.
// ---------------------------------------------------------------------------
__global__ __launch_bounds__(256) void dense_gemm_kernel(
    const __bf16* __restrict__ A, const float* __restrict__ W,
    const float* __restrict__ bias, float* __restrict__ out)
{
  __shared__ alignas(16) __bf16 As[128 * 72];
  __shared__ alignas(16) __bf16 Bs[128 * 72];
  const int tid = threadIdx.x, bm = blockIdx.x, bn = blockIdx.y;
  const int w = tid >> 5, lane = tid & 31, half = lane >> 4, l16 = lane & 15;
  const int wr = w >> 2, wc = w & 3;

  v8f acc[4][2];
#pragma unroll
  for (int i = 0; i < 4; ++i)
#pragma unroll
    for (int j = 0; j < 2; ++j) acc[i][j] = zero8();

  for (int k0 = 0; k0 < H_DIM; k0 += 64) {
    __syncthreads();
#pragma unroll
    for (int c = tid; c < 512; c += 256) {
      int row = c >> 2, kc = (c & 3) * 16;
      stage_async32B(A + (size_t)(bm * 128 + row) * H_DIM + k0 + kc,
                     &As[row * 72 + kc]);
    }
#pragma unroll
    for (int c = tid; c < 512; c += 256) {
      int row = c >> 2, kc = (c & 3) * 16;
      stage_f32(W + (size_t)(bn * 128 + row) * H_DIM + k0 + kc, &Bs[row * 72 + kc]);
    }
    if (k0 + 64 < H_DIM)
      __builtin_prefetch(W + (size_t)(bn * 128 + (tid >> 1)) * H_DIM + k0 + 64, 0, 1);
    wait_async0();
    __syncthreads();
#pragma unroll
    for (int kk = 0; kk < 64; kk += 32) {
      v16bf af[4];
#pragma unroll
      for (int i = 0; i < 4; ++i)
        af[i] = ld_afrag(&As[(wr * 64 + i * 16 + l16) * 72], kk, half);
      v16bf bfv[2];
#pragma unroll
      for (int j = 0; j < 2; ++j)
        bfv[j] = ld_bfrag(&Bs[(wc * 32 + j * 16 + l16) * 72], kk, half);
#pragma unroll
      for (int i = 0; i < 4; ++i)
#pragma unroll
        for (int j = 0; j < 2; ++j)
          acc[i][j] = wmma_bf16(af[i], bfv[j], acc[i][j]);
    }
  }

#pragma unroll
  for (int j = 0; j < 2; ++j) {
    const int n = bn * 128 + wc * 32 + j * 16 + l16;
    const float bv = bias[n];
#pragma unroll
    for (int i = 0; i < 4; ++i) {
#pragma unroll
      for (int v = 0; v < 8; ++v) {
        const int srow = bm * 128 + wr * 64 + i * 16 + half * 8 + v;
        out[(size_t)srow * H_DIM + n] = acc[i][j][v] + bv;
      }
    }
  }
}

// ---------------------------------------------------------------------------
// Launch: QKV GEMM -> RoPE -> flash attention -> dense GEMM.
// Workspace: Q,K,V,ctx bf16 = 4 * 16 MB = 64 MB.
// ---------------------------------------------------------------------------
extern "C" void kernel_launch(void* const* d_in, const int* in_sizes, int n_in,
                              void* d_out, int out_size, void* d_ws, size_t ws_size,
                              hipStream_t stream) {
  (void)in_sizes; (void)n_in; (void)out_size; (void)ws_size;
  const float* hidden  = (const float*)d_in[0];
  // d_in[1] = attention_mask (causal tril) -- causality is hard-coded.
  const float* W_qkv   = (const float*)d_in[2];
  const float* b_qkv   = (const float*)d_in[3];
  const float* W_dense = (const float*)d_in[4];
  const float* b_dense = (const float*)d_in[5];
  float* out = (float*)d_out;

  const size_t perBuf = (size_t)NHEAD * S_LEN * HDIM;   // 8M bf16 elems
  __bf16* Qb  = (__bf16*)d_ws;
  __bf16* Kb  = Qb + perBuf;
  __bf16* Vb  = Kb + perBuf;
  __bf16* Ctx = Vb + perBuf;

  qkv_gemm_kernel<<<dim3(S_LEN / 128, (3 * H_DIM) / 128), 256, 0, stream>>>(
      hidden, W_qkv, b_qkv, Qb, Kb, Vb);
  rope_kernel<<<(S_LEN * NHEAD * 16) / 256, 256, 0, stream>>>(Qb, Kb);
  attn_kernel<<<dim3(S_LEN / 128, NHEAD), 256, 0, stream>>>(Qb, Kb, Vb, Ctx);
  dense_gemm_kernel<<<dim3(S_LEN / 128, H_DIM / 128), 256, 0, stream>>>(
      Ctx, W_dense, b_dense, out);
}
